// ProteinGCN_11957188952167
// MI455X (gfx1250) — compile-verified, matrix operand order checked by reading, not run on hardware
//
#include <hip/hip_runtime.h>
#include <hip/hip_bf16.h>

// ---------------------------------------------------------------------------
// ProteinGCN pipeline for MI455X (gfx1250, wave32, WMMA)
//   3x [ GEMM(bf16 WMMA frag-layout, f32 acc, double-buffered LDS A-staging)
//        -> src-scale -> edge scatter-add -> dst-scale + bias (+relu) ]
//   -> graph mean pool -> LayerNorm
// ---------------------------------------------------------------------------

#define N_NODES 50000
#define N_EDGES 800000
#define N_GRAPHS 50
#define IN_FEATS 512
#define HID 128

typedef __bf16 v16bf __attribute__((ext_vector_type(16)));
typedef float  v8f   __attribute__((ext_vector_type(8)));

__device__ __forceinline__ __bf16 to_bf16(float x) { return (__bf16)x; }

// ---------------------------------------------------------------------------
// Zero fill (float4 granularity)
// ---------------------------------------------------------------------------
__global__ void zero_f32(float* __restrict__ p, int n4) {
    int i = blockIdx.x * blockDim.x + threadIdx.x;
    if (i < n4) {
        float4 z; z.x = 0.f; z.y = 0.f; z.z = 0.f; z.w = 0.f;
        ((float4*)p)[i] = z;
    }
}

// ---------------------------------------------------------------------------
// Degree accumulation: deg_out[src[e]] += 1 ; deg_in[dst[e]] += 1
// ---------------------------------------------------------------------------
__global__ void degree_kernel(const int* __restrict__ src, const int* __restrict__ dst,
                              float* __restrict__ degOut, float* __restrict__ degIn, int nE) {
    int e = blockIdx.x * blockDim.x + threadIdx.x;
    if (e < nE) {
        atomicAdd(&degOut[src[e]], 1.0f);
        atomicAdd(&degIn[dst[e]], 1.0f);
    }
}

// in-place: deg -> rsqrt(max(deg,1))
__global__ void inv_sqrt_kernel(float* __restrict__ degOut, float* __restrict__ degIn, int n) {
    int i = blockIdx.x * blockDim.x + threadIdx.x;
    if (i < n) {
        degOut[i] = rsqrtf(fmaxf(degOut[i], 1.0f));
        degIn[i]  = rsqrtf(fmaxf(degIn[i],  1.0f));
    }
}

// ---------------------------------------------------------------------------
// Pre-pack W[K,128] (f32) into bf16 WMMA B-fragment order (wave32 ISA 7.12.2):
//   fragment f = (kstep*8 + wave)*32 + lane ; 16 bf16 per fragment.
//   lane = (col&15) + ((k>>3)&1)*16 ; elem i = (k&7) + (k>=16 ? 8 : 0)
// One thread per (fragment, element-pair): pairs never cross an octet.
// ---------------------------------------------------------------------------
__global__ void pack_w_bf16(const float* __restrict__ W, __bf16* __restrict__ Wf, int K) {
    int t = blockIdx.x * blockDim.x + threadIdx.x;
    int total = (K / 32) * 8 * 32 * 8;      // frags * 8 pairs
    if (t >= total) return;
    int ip   = t & 7;                       // pair -> i0 = 2*ip
    int f    = t >> 3;
    int lane = f & 31;
    int w    = (f >> 5) & 7;
    int s    = f >> 8;
    int kb   = (lane >> 4) * 8;
    int n    = lane & 15;
    int i0   = 2 * ip;
    int k0   = kb + (i0 < 8 ? i0 : 16 + (i0 - 8));   // octet-local, k1 = k0+1
    float x0 = W[(s * 32 + k0) * 128 + w * 16 + n];
    float x1 = W[(s * 32 + k0 + 1) * 128 + w * 16 + n];
    __bf16* o = Wf + ((size_t)f * 16 + i0);
    o[0] = to_bf16(x0);
    o[1] = to_bf16(x1);
}

// ---------------------------------------------------------------------------
// GEMM: Y[N,128] = (X[N,K] @ W[K,128]) * rowScale[row]
// block = 256 threads (8 waves) -> 16 rows x 128 cols; wave w owns cols [16w,16w+16).
// A tile staged in LDS in fragment order, DOUBLE-BUFFERED: one barrier per
// k-step (write buf[(s+1)&1] while reading buf[s&1]; reuse distance = 2 barriers).
// B fragments streamed from pre-packed L2-resident global buffer (b128 loads).
// K templated -> fully unrolled -> 16 (K=512) / 4 (K=128) v_wmma per wave.
// ---------------------------------------------------------------------------
template <int K>
__global__ void __launch_bounds__(256)
gemm_wmma_bf16(const float* __restrict__ X, const __bf16* __restrict__ Wf,
               const float* __restrict__ rowScale, float* __restrict__ Y) {
    const int tid  = threadIdx.x;
    const int wave = tid >> 5;
    const int lane = tid & 31;
    const int rowBase = blockIdx.x * 16;
    constexpr int NSTEP = K / 32;

    // A fragments, double buffered: 2 x 32 lanes x 16 bf16 (32B/lane, aligned)
    __shared__ __bf16 Af[2][32][16];

    // A staging assignment: thread t handles row pr, k-pair pk (same octet)
    const int pr = tid >> 4;            // 0..15
    const int pk = (tid & 15) * 2;      // 0,2,..,30
    const int plane = pr + ((pk >> 3) & 1) * 16;
    const int pi    = (pk & 7) + (pk >= 16 ? 8 : 0);
    const float* xrow = X + (size_t)(rowBase + pr) * K;

    // prologue: stage step 0
    {
        float2 x = *(const float2*)(xrow + pk);
        Af[0][plane][pi]     = to_bf16(x.x);
        Af[0][plane][pi + 1] = to_bf16(x.y);
    }
    __syncthreads();

    v8f acc = {};
#pragma unroll
    for (int s = 0; s < NSTEP; ++s) {
        // prefetch next step's A tile into the other buffer (no conflict)
        if (s + 1 < NSTEP) {
            float2 x = *(const float2*)(xrow + (s + 1) * 32 + pk);
            Af[(s + 1) & 1][plane][pi]     = to_bf16(x.x);
            Af[(s + 1) & 1][plane][pi + 1] = to_bf16(x.y);
        }

        v16bf a = *(const v16bf*)(&Af[s & 1][lane][0]);
        v16bf b = *(const v16bf*)(Wf + ((size_t)(s * 8 + wave) * 32 + lane) * 16);

        acc = __builtin_amdgcn_wmma_f32_16x16x32_bf16(
            /*neg_a=*/false, a, /*neg_b=*/false, b,
            /*c_mod=*/(short)0, acc, /*reuse_a=*/false, /*reuse_b=*/false);

        __syncthreads();   // publish buf[(s+1)&1]; reads of buf[s&1] are done
    }

    // C/D layout: VGPR j holds row j (lanes 0-15) / row j+8 (lanes 16-31)
    const int n    = lane & 15;
    const int rOff = (lane >> 4) * 8;
#pragma unroll
    for (int j = 0; j < 8; ++j) {
        int row = rowBase + rOff + j;
        Y[(size_t)row * 128 + wave * 16 + n] = acc[j] * rowScale[row];
    }
}

// ---------------------------------------------------------------------------
// SpMM scatter: M[dst[e], :] += Y[src[e], :]   (one wave per edge, float4/lane)
// 25.6 MB node matrix is L2-resident -> atomics execute at L2.
// ---------------------------------------------------------------------------
__global__ void __launch_bounds__(256)
spmm_scatter(const float* __restrict__ Y, const int* __restrict__ src,
             const int* __restrict__ dst, float* __restrict__ M, int nE) {
    int gw = (int)((blockIdx.x * blockDim.x + threadIdx.x) >> 5);
    int lane = threadIdx.x & 31;
    if (gw >= nE) return;
    int s = src[gw];
    int d = dst[gw];
    const float4 v = *(const float4*)(Y + (size_t)s * 128 + lane * 4);
    float* o = M + (size_t)d * 128 + lane * 4;
    atomicAdd(o + 0, v.x);
    atomicAdd(o + 1, v.y);
    atomicAdd(o + 2, v.z);
    atomicAdd(o + 3, v.w);
}

// ---------------------------------------------------------------------------
// Node epilogue: H = (relu?) (M * dinScale[row] + bias[col])   (float4/thread)
// ---------------------------------------------------------------------------
__global__ void node_epilogue(const float* __restrict__ M, const float* __restrict__ dinScale,
                              const float* __restrict__ bias, float* __restrict__ H,
                              int doRelu) {
    int i = blockIdx.x * blockDim.x + threadIdx.x;      // float4 index
    if (i >= N_NODES * 32) return;
    int node = i >> 5;
    int c4   = (i & 31) * 4;
    float s = dinScale[node];
    float4 m = *(const float4*)(M + (size_t)node * 128 + c4);
    float4 b = *(const float4*)(bias + c4);
    float4 h;
    h.x = fmaf(m.x, s, b.x);
    h.y = fmaf(m.y, s, b.y);
    h.z = fmaf(m.z, s, b.z);
    h.w = fmaf(m.w, s, b.w);
    if (doRelu) {
        h.x = fmaxf(h.x, 0.f); h.y = fmaxf(h.y, 0.f);
        h.z = fmaxf(h.z, 0.f); h.w = fmaxf(h.w, 0.f);
    }
    *(float4*)(H + (size_t)node * 128 + c4) = h;
}

// ---------------------------------------------------------------------------
// Graph mean-pool part 1: per-graph sums + counts (wave per node)
// ---------------------------------------------------------------------------
__global__ void __launch_bounds__(256)
pool_kernel(const float* __restrict__ H, const int* __restrict__ gid,
            float* __restrict__ sums, float* __restrict__ counts) {
    int node = (int)((blockIdx.x * blockDim.x + threadIdx.x) >> 5);
    int lane = threadIdx.x & 31;
    if (node >= N_NODES) return;
    int g = gid[node];
    float4 v = *(const float4*)(H + (size_t)node * 128 + lane * 4);
    float* o = sums + (size_t)g * 128 + lane * 4;
    atomicAdd(o + 0, v.x);
    atomicAdd(o + 1, v.y);
    atomicAdd(o + 2, v.z);
    atomicAdd(o + 3, v.w);
    if (lane == 0) atomicAdd(&counts[g], 1.0f);
}

// ---------------------------------------------------------------------------
// Mean + LayerNorm: one 128-thread block per graph
// ---------------------------------------------------------------------------
__global__ void __launch_bounds__(128)
ln_kernel(const float* __restrict__ sums, const float* __restrict__ counts,
          const float* __restrict__ gamma, const float* __restrict__ beta,
          float* __restrict__ out) {
    int g = blockIdx.x;
    int c = threadIdx.x;
    float cnt = fmaxf(counts[g], 1.0f);
    float v = sums[g * 128 + c] / cnt;

    __shared__ float red[128];
    red[c] = v;
    __syncthreads();
    for (int s = 64; s > 0; s >>= 1) {
        if (c < s) red[c] += red[c + s];
        __syncthreads();
    }
    float mu = red[0] * (1.0f / 128.0f);
    __syncthreads();
    float d = v - mu;
    red[c] = d * d;
    __syncthreads();
    for (int s = 64; s > 0; s >>= 1) {
        if (c < s) red[c] += red[c + s];
        __syncthreads();
    }
    float var = red[0] * (1.0f / 128.0f);
    float r = rsqrtf(var + 1e-5f);
    out[g * 128 + c] = d * r * gamma[c] + beta[c];
}

// ---------------------------------------------------------------------------
// Orchestration (all on `stream`, graph-capture safe)
// ---------------------------------------------------------------------------
extern "C" void kernel_launch(void* const* d_in, const int* in_sizes, int n_in,
                              void* d_out, int out_size, void* d_ws, size_t ws_size,
                              hipStream_t stream) {
    (void)in_sizes; (void)n_in; (void)out_size; (void)ws_size;

    const float* features = (const float*)d_in[0];   // [50000,512]
    const int*   src      = (const int*)  d_in[1];   // [800000]
    const int*   dst      = (const int*)  d_in[2];   // [800000]
    const int*   gid      = (const int*)  d_in[3];   // [50000]
    const float* W1 = (const float*)d_in[4];
    const float* b1 = (const float*)d_in[5];
    const float* W2 = (const float*)d_in[6];
    const float* b2 = (const float*)d_in[7];
    const float* W3 = (const float*)d_in[8];
    const float* b3 = (const float*)d_in[9];
    const float* gamma = (const float*)d_in[10];
    const float* beta  = (const float*)d_in[11];

    float* out_hg = (float*)d_out;                   // [50,128]
    float* out_h  = (float*)d_out + N_GRAPHS * HID;  // [50000,128]

    // workspace layout (floats, then bf16 region)
    float* ws   = (float*)d_ws;
    float* buf0 = ws;                                // 50000*128
    float* buf1 = buf0 + (size_t)N_NODES * HID;      // 50000*128
    float* invDout = buf1 + (size_t)N_NODES * HID;   // 50000
    float* invDin  = invDout + N_NODES;              // 50000
    float* gsums   = invDin + N_NODES;               // 50*128
    float* gcounts = gsums + N_GRAPHS * HID;         // 50 (+ pad to 32B)
    __bf16* Wf1 = (__bf16*)(gcounts + 64);           // 512*128 bf16
    __bf16* Wf2 = Wf1 + (size_t)IN_FEATS * HID;      // 128*128 bf16
    __bf16* Wf3 = Wf2 + (size_t)HID * HID;           // 128*128 bf16

    const int nodeF4  = N_NODES * 32;                // node matrix in float4s
    dim3 b256(256), b128(128);
    dim3 gZeroNode((nodeF4 + 255) / 256);
    dim3 gGemm(N_NODES / 16);                        // 3125 (exact)
    dim3 gEdgeWave(N_EDGES / 8);                     // wave/edge, 8 waves/block
    dim3 gEdgeThr((N_EDGES + 255) / 256);
    dim3 gNodeWave(N_NODES / 8 + 1);                 // wave/node
    dim3 gNodeThr((N_NODES + 255) / 256);

    // ---- pre-pack weights to bf16 fragment layout (L2-resident B operands) ----
    {
        int t1 = (IN_FEATS / 32) * 8 * 32 * 8;       // 32768
        int t2 = (HID / 32) * 8 * 32 * 8;            // 8192
        pack_w_bf16<<<(t1 + 255) / 256, b256, 0, stream>>>(W1, Wf1, IN_FEATS);
        pack_w_bf16<<<(t2 + 255) / 256, b256, 0, stream>>>(W2, Wf2, HID);
        pack_w_bf16<<<(t2 + 255) / 256, b256, 0, stream>>>(W3, Wf3, HID);
    }

    // ---- degrees -> inv-sqrt scales ----
    {
        int degF4 = (2 * N_NODES) / 4;               // invDout & invDin contiguous
        zero_f32<<<(degF4 + 255) / 256, b256, 0, stream>>>(invDout, degF4);
        degree_kernel<<<gEdgeThr, b256, 0, stream>>>(src, dst, invDout, invDin, N_EDGES);
        inv_sqrt_kernel<<<gNodeThr, b256, 0, stream>>>(invDout, invDin, N_NODES);
    }

    // ---- layer 1: features(K=512) -> buf0 -> scatter buf1 -> H1 in buf0 ----
    gemm_wmma_bf16<IN_FEATS><<<gGemm, b256, 0, stream>>>(features, Wf1, invDout, buf0);
    zero_f32<<<gZeroNode, b256, 0, stream>>>(buf1, nodeF4);
    spmm_scatter<<<gEdgeWave, b256, 0, stream>>>(buf0, src, dst, buf1, N_EDGES);
    node_epilogue<<<gZeroNode, b256, 0, stream>>>(buf1, invDin, b1, buf0, /*relu=*/1);

    // ---- layer 2: buf0(K=128) -> buf1 -> scatter buf0 -> H2 in buf1 ----
    gemm_wmma_bf16<HID><<<gGemm, b256, 0, stream>>>(buf0, Wf2, invDout, buf1);
    zero_f32<<<gZeroNode, b256, 0, stream>>>(buf0, nodeF4);
    spmm_scatter<<<gEdgeWave, b256, 0, stream>>>(buf1, src, dst, buf0, N_EDGES);
    node_epilogue<<<gZeroNode, b256, 0, stream>>>(buf0, invDin, b2, buf1, /*relu=*/1);

    // ---- layer 3: buf1(K=128) -> buf0 -> scatter buf1 -> h into d_out ----
    gemm_wmma_bf16<HID><<<gGemm, b256, 0, stream>>>(buf1, Wf3, invDout, buf0);
    zero_f32<<<gZeroNode, b256, 0, stream>>>(buf1, nodeF4);
    spmm_scatter<<<gEdgeWave, b256, 0, stream>>>(buf0, src, dst, buf1, N_EDGES);
    node_epilogue<<<gZeroNode, b256, 0, stream>>>(buf1, invDin, b3, out_h, /*relu=*/0);

    // ---- graph mean pool + LayerNorm -> hg into d_out ----
    {
        int pf4 = (N_GRAPHS * HID + N_GRAPHS + 3) / 4 + 1; // sums + counts
        zero_f32<<<(pf4 + 255) / 256, b256, 0, stream>>>(gsums, pf4);
        pool_kernel<<<gNodeWave, b256, 0, stream>>>(out_h, gid, gsums, gcounts);
        ln_kernel<<<N_GRAPHS, b128, 0, stream>>>(gsums, gcounts, gamma, beta, out_hg);
    }
}
